// Net_9320079033153
// MI455X (gfx1250) — compile-verified, matrix operand order checked by reading, not run on hardware
//
#include <hip/hip_runtime.h>

#define N_NODES   100000
#define F_IN      512
#define HID       512
#define NCLS      40
#define NCLS_PAD  48
#define NEDGE     1600000
#define KITER     10
#define ALPHA     0.1f

typedef __bf16 bf16_t;
typedef __attribute__((ext_vector_type(16))) __bf16 v16bf;
typedef __attribute__((ext_vector_type(8)))  __bf16 v8bf;
typedef __attribute__((ext_vector_type(4)))  __bf16 v4bf;
typedef __attribute__((ext_vector_type(8)))  float  v8f;

// ---------------------------------------------------------------------------
// Fragment loader: 16x32 bf16 tile, K-contiguous (row-major over K).
// Per CDNA5 ISA 16-bit A/B layout: lanes 0-15 hold K[k0..k0+7] in VGPR0-3 and
// K[k0+16..k0+23] in VGPR4-7; lanes 16-31 hold K[k0+8..15] / K[k0+24..31].
// => two coalesced 16-byte loads per lane.
// ---------------------------------------------------------------------------
__device__ __forceinline__ v16bf load_frag16x32(const bf16_t* __restrict__ rowbase,
                                                int k0, int hf) {
  const bf16_t* p = rowbase + k0 + hf * 8;
  v8bf lo = *(const v8bf*)(p);
  v8bf hi = *(const v8bf*)(p + 16);
  v16bf f;
#pragma unroll
  for (int i = 0; i < 8; ++i) { f[i] = lo[i]; f[i + 8] = hi[i]; }
  return f;
}

// ---------------------------------------------------------------------------
// GEMM1: H = relu(Xbf @ W1t^T + b1).   Xbf [N][512] bf16, W1t [512][512] bf16
// (N-major, K contiguous).  Block = 8 waves; block owns a 16-row M tile; wave
// w owns 64 columns (4 accumulators).  K loop unrolled x2 with two statically
// named fragment buffer sets -> copy-free software pipeline: loads for one
// parity stay in flight across the other parity's WMMAs.
// ---------------------------------------------------------------------------
__global__ __launch_bounds__(256) void gemm1_relu_bf16(
    const bf16_t* __restrict__ A,
    const bf16_t* __restrict__ Bt,
    const float*  __restrict__ bias,
    bf16_t* __restrict__ H)
{
  const int lane = threadIdx.x & 31;
  const int wv   = threadIdx.x >> 5;       // 0..7
  const int hf   = lane >> 4;
  const int ln   = lane & 15;
  const int m0   = blockIdx.x << 4;
  const int n0   = wv << 6;

  const bf16_t* arow = A + (size_t)(m0 + ln) * F_IN;
  const bf16_t* brow[4];
#pragma unroll
  for (int j = 0; j < 4; ++j) brow[j] = Bt + (size_t)(n0 + j * 16 + ln) * F_IN;

  v8f acc[4] = {};

  // parity-A buffers, preloaded at k=0
  v16bf aA = load_frag16x32(arow, 0, hf);
  v16bf bA[4];
#pragma unroll
  for (int j = 0; j < 4; ++j) bA[j] = load_frag16x32(brow[j], 0, hf);

  for (int k0 = 0; k0 < F_IN; k0 += 64) {
    // prefetch parity-B (k0+32)
    const int kB = k0 + 32;
    v16bf aB = load_frag16x32(arow, kB, hf);
    v16bf bB[4];
#pragma unroll
    for (int j = 0; j < 4; ++j) bB[j] = load_frag16x32(brow[j], kB, hf);

    // compute parity-A (k0)
#pragma unroll
    for (int j = 0; j < 4; ++j)
      acc[j] = __builtin_amdgcn_wmma_f32_16x16x32_bf16(
          false, aA, false, bA[j], (short)0, acc[j], false, false);

    // prefetch parity-A for next iteration (k0+64, wraps harmlessly at end)
    const int kN = (k0 + 64) & (F_IN - 1);
    aA = load_frag16x32(arow, kN, hf);
#pragma unroll
    for (int j = 0; j < 4; ++j) bA[j] = load_frag16x32(brow[j], kN, hf);

    // compute parity-B (k0+32)
#pragma unroll
    for (int j = 0; j < 4; ++j)
      acc[j] = __builtin_amdgcn_wmma_f32_16x16x32_bf16(
          false, aB, false, bB[j], (short)0, acc[j], false, false);
  }

#pragma unroll
  for (int j = 0; j < 4; ++j) {
    const int col = n0 + j * 16 + ln;
    const float bv = bias[col];
#pragma unroll
    for (int i = 0; i < 8; ++i) {
      const int row = m0 + i + hf * 8;
      float v = acc[j][i] + bv;
      v = v > 0.0f ? v : 0.0f;
      H[(size_t)row * HID + col] = (bf16_t)v;
    }
  }
}

// ---------------------------------------------------------------------------
// GEMM2: Z0 = H @ W2t^T + b2.  One wave per 16-row tile, 3 n-tiles (48 cols,
// columns >= 40 are padding and not stored).  Same copy-free pipelining.
// ---------------------------------------------------------------------------
__global__ __launch_bounds__(32) void gemm2_bf16(
    const bf16_t* __restrict__ H,
    const bf16_t* __restrict__ Bt,    // [48][512]
    const float*  __restrict__ bias,  // [40]
    float* __restrict__ Z)            // [N][40]
{
  const int lane = threadIdx.x & 31;
  const int hf   = lane >> 4;
  const int ln   = lane & 15;
  const int m0   = blockIdx.x << 4;

  const bf16_t* arow = H + (size_t)(m0 + ln) * HID;
  const bf16_t* brow[3];
#pragma unroll
  for (int j = 0; j < 3; ++j) brow[j] = Bt + (size_t)(j * 16 + ln) * HID;

  v8f acc[3] = {};

  v16bf aA = load_frag16x32(arow, 0, hf);
  v16bf bA[3];
#pragma unroll
  for (int j = 0; j < 3; ++j) bA[j] = load_frag16x32(brow[j], 0, hf);

  for (int k0 = 0; k0 < HID; k0 += 64) {
    const int kB = k0 + 32;
    v16bf aB = load_frag16x32(arow, kB, hf);
    v16bf bB[3];
#pragma unroll
    for (int j = 0; j < 3; ++j) bB[j] = load_frag16x32(brow[j], kB, hf);

#pragma unroll
    for (int j = 0; j < 3; ++j)
      acc[j] = __builtin_amdgcn_wmma_f32_16x16x32_bf16(
          false, aA, false, bA[j], (short)0, acc[j], false, false);

    const int kN = (k0 + 64) & (HID - 1);
    aA = load_frag16x32(arow, kN, hf);
#pragma unroll
    for (int j = 0; j < 3; ++j) bA[j] = load_frag16x32(brow[j], kN, hf);

#pragma unroll
    for (int j = 0; j < 3; ++j)
      acc[j] = __builtin_amdgcn_wmma_f32_16x16x32_bf16(
          false, aB, false, bB[j], (short)0, acc[j], false, false);
  }

#pragma unroll
  for (int j = 0; j < 3; ++j) {
    const int col = j * 16 + ln;
    if (col < NCLS) {
      const float bv = bias[col];
#pragma unroll
      for (int i = 0; i < 8; ++i) {
        const int row = m0 + i + hf * 8;
        Z[(size_t)row * NCLS + col] = acc[j][i] + bv;
      }
    }
  }
}

// ------------------------- packing / conversion ----------------------------
__global__ void cvt_f32_bf16(const float* __restrict__ in, bf16_t* __restrict__ out, int n4) {
  int i = blockIdx.x * blockDim.x + threadIdx.x;
  if (i >= n4) return;
  float4 v = ((const float4*)in)[i];
  v4bf o;
  o[0] = (bf16_t)v.x; o[1] = (bf16_t)v.y; o[2] = (bf16_t)v.z; o[3] = (bf16_t)v.w;
  *(v4bf*)(out + (size_t)i * 4) = o;
}

__global__ void pack_w1t(const float* __restrict__ W, bf16_t* __restrict__ Wt) {
  int idx = blockIdx.x * blockDim.x + threadIdx.x;
  if (idx >= HID * F_IN) return;
  int n = idx / F_IN, k = idx % F_IN;          // Wt[n][k] = W[k][n]
  Wt[idx] = (bf16_t)W[(size_t)k * HID + n];
}

__global__ void pack_w2t(const float* __restrict__ W, bf16_t* __restrict__ Wt) {
  int idx = blockIdx.x * blockDim.x + threadIdx.x;
  if (idx >= NCLS_PAD * HID) return;
  int n = idx / HID, k = idx % HID;            // Wt[n][k] = W[k][n], zero-padded
  float v = (n < NCLS) ? W[(size_t)k * NCLS + n] : 0.0f;
  Wt[idx] = (bf16_t)v;
}

// ----------------------------- APPNP kernels -------------------------------
__global__ void zero_deg(int* __restrict__ deg) {
  int i = blockIdx.x * blockDim.x + threadIdx.x;
  if (i < N_NODES) deg[i] = 0;
}

__global__ void count_deg(const int* __restrict__ row, int* __restrict__ deg) {
  int e = blockIdx.x * blockDim.x + threadIdx.x;
  if (e < NEDGE) atomicAdd(&deg[row[e]], 1);
}

__global__ void calc_dinv(const int* __restrict__ deg, float* __restrict__ dinv) {
  int i = blockIdx.x * blockDim.x + threadIdx.x;
  if (i < N_NODES) dinv[i] = rsqrtf((float)(deg[i] + 1));   // +1 self-loop
}

// Edge weights are identical across all K iterations: compute once.
__global__ void calc_wedge(const int* __restrict__ row, const int* __restrict__ col,
                           const float* __restrict__ dinv, float* __restrict__ wedge) {
  int e = blockIdx.x * blockDim.x + threadIdx.x;
  if (e < NEDGE) wedge[e] = dinv[row[e]] * dinv[col[e]];
}

// agg = dinv[i]^2 * z[i]   (self-loop term folded into the init)
__global__ void init_agg(const float* __restrict__ z, const float* __restrict__ dinv,
                         float* __restrict__ agg) {
  int idx = blockIdx.x * blockDim.x + threadIdx.x;
  if (idx >= N_NODES * (NCLS / 4)) return;
  int node = idx / (NCLS / 4);
  float s = dinv[node];
  float w = s * s;
  float4 v = ((const float4*)z)[idx];
  float4 o; o.x = w * v.x; o.y = w * v.y; o.z = w * v.z; o.w = w * v.w;
  ((float4*)agg)[idx] = o;
}

__global__ void edge_scatter(const float* __restrict__ z, const float* __restrict__ wedge,
                             const int* __restrict__ row, const int* __restrict__ col,
                             float* __restrict__ agg) {
  int idx = blockIdx.x * blockDim.x + threadIdx.x;
  if (idx >= NEDGE * (NCLS / 4)) return;
  int e = idx / (NCLS / 4);
  int q = idx - e * (NCLS / 4);
  int r = row[e];
  int c = col[e];
  float w = wedge[e];
  float4 v = ((const float4*)(z + (size_t)c * NCLS))[q];
  float* dst = agg + (size_t)r * NCLS + q * 4;
  unsafeAtomicAdd(dst + 0, w * v.x);
  unsafeAtomicAdd(dst + 1, w * v.y);
  unsafeAtomicAdd(dst + 2, w * v.z);
  unsafeAtomicAdd(dst + 3, w * v.w);
}

__global__ void appnp_combine(const float* __restrict__ agg, const float* __restrict__ z0,
                              float* __restrict__ out) {
  int idx = blockIdx.x * blockDim.x + threadIdx.x;
  if (idx >= N_NODES * (NCLS / 4)) return;
  float4 a = ((const float4*)agg)[idx];
  float4 z = ((const float4*)z0)[idx];
  float4 o;
  o.x = (1.0f - ALPHA) * a.x + ALPHA * z.x;
  o.y = (1.0f - ALPHA) * a.y + ALPHA * z.y;
  o.z = (1.0f - ALPHA) * a.z + ALPHA * z.z;
  o.w = (1.0f - ALPHA) * a.w + ALPHA * z.w;
  ((float4*)out)[idx] = o;
}

// ---------------------------------------------------------------------------
extern "C" void kernel_launch(void* const* d_in, const int* in_sizes, int n_in,
                              void* d_out, int out_size, void* d_ws, size_t ws_size,
                              hipStream_t stream) {
  (void)in_sizes; (void)n_in; (void)out_size; (void)ws_size;
  const float* x    = (const float*)d_in[0];
  const float* W1   = (const float*)d_in[1];
  const float* b1   = (const float*)d_in[2];
  const float* W2   = (const float*)d_in[3];
  const float* b2   = (const float*)d_in[4];
  const int*   eidx = (const int*)d_in[5];
  const int*   erow = eidx;           // targets
  const int*   ecol = eidx + NEDGE;   // sources

  char* ws = (char*)d_ws;
  size_t off = 0;
  auto take = [&](size_t bytes) -> char* {
    char* p = ws + off;
    off += (bytes + 255) & ~(size_t)255;
    return p;
  };
  bf16_t* xbf   = (bf16_t*)take((size_t)N_NODES * F_IN * 2);     // 102.4 MB
  bf16_t* hbf   = (bf16_t*)take((size_t)N_NODES * HID * 2);      // 102.4 MB
  bf16_t* w1t   = (bf16_t*)take((size_t)HID * F_IN * 2);         // 0.5 MB
  bf16_t* w2t   = (bf16_t*)take((size_t)NCLS_PAD * HID * 2);     // 48 KB
  float*  z0    = (float*)take((size_t)N_NODES * NCLS * 4);      // 16 MB
  float*  ztmp  = (float*)take((size_t)N_NODES * NCLS * 4);      // 16 MB
  float*  agg   = (float*)take((size_t)N_NODES * NCLS * 4);      // 16 MB
  float*  wedge = (float*)take((size_t)NEDGE * 4);               // 6.4 MB
  int*    deg   = (int*)take((size_t)N_NODES * 4);
  float*  dinv  = (float*)take((size_t)N_NODES * 4);

  const int TB = 256;

  // --- pack inputs to bf16 / transposed weights ---
  cvt_f32_bf16<<<(N_NODES * F_IN / 4 + TB - 1) / TB, TB, 0, stream>>>(x, xbf, N_NODES * F_IN / 4);
  pack_w1t<<<(HID * F_IN + TB - 1) / TB, TB, 0, stream>>>(W1, w1t);
  pack_w2t<<<(NCLS_PAD * HID + TB - 1) / TB, TB, 0, stream>>>(W2, w2t);

  // --- degrees / normalization / edge weights ---
  zero_deg<<<(N_NODES + TB - 1) / TB, TB, 0, stream>>>(deg);
  count_deg<<<(NEDGE + TB - 1) / TB, TB, 0, stream>>>(erow, deg);
  calc_dinv<<<(N_NODES + TB - 1) / TB, TB, 0, stream>>>(deg, dinv);
  calc_wedge<<<(NEDGE + TB - 1) / TB, TB, 0, stream>>>(erow, ecol, dinv, wedge);

  // --- MLP via WMMA ---
  gemm1_relu_bf16<<<N_NODES / 16, 256, 0, stream>>>(xbf, w1t, b1, hbf);
  gemm2_bf16<<<N_NODES / 16, 32, 0, stream>>>(hbf, w2t, b2, z0);

  // --- APPNP propagation, ping-pong so iteration 9 lands in d_out ---
  const int nvec  = N_NODES * (NCLS / 4);
  const int evec  = NEDGE * (NCLS / 4);
  const int gN    = (nvec + TB - 1) / TB;
  const int gE    = (evec + TB - 1) / TB;

  const float* zc = z0;
  float* zout = (float*)d_out;
  for (int t = 0; t < KITER; ++t) {
    float* dst = (t % 2 == 0) ? ztmp : zout;
    init_agg<<<gN, TB, 0, stream>>>(zc, dinv, agg);
    edge_scatter<<<gE, TB, 0, stream>>>(zc, wedge, erow, ecol, agg);
    appnp_combine<<<gN, TB, 0, stream>>>(agg, z0, dst);
    zc = dst;
  }
}
